// ReportDecoder_9448928051412
// MI455X (gfx1250) — matrix operand order repaired and checked by reading, not compiled, not run on hardware
//
#include <hip/hip_runtime.h>

#define BB 256
#define PP 49
#define EE 1024
#define HH 1024
#define VV 8
#define LL 128
#define TT 32
#define IN_ 1161
#define G4 4096      // 4*H
#define NO 1154      // L+H+2
#define NOP 1168     // NO padded to multiple of 16

typedef __attribute__((ext_vector_type(16))) __bf16        v16bf;
typedef __attribute__((ext_vector_type(8)))  float         v8f;
typedef __attribute__((ext_vector_type(4)))  unsigned int  u32x4;

union ABu { u32x4 u[2]; v16bf v; };

__device__ __forceinline__ unsigned short f2bf(float f) {
    unsigned int u = __builtin_bit_cast(unsigned int, f);
    unsigned int lsb = (u >> 16) & 1u;
    u += 0x7fffu + lsb;                 // round-to-nearest-even
    return (unsigned short)(u >> 16);
}
__device__ __forceinline__ float sigm(float x) { return 1.0f / (1.0f + expf(-x)); }

// A fragment: 16x32 bf16, A row-major (lda in elements). contiguous k order used here (same as B):
// lane<16 : row=lane,   K 0..15
// lane>=16: row=lane-16,K 16..31
__device__ __forceinline__ v16bf load_a_frag(const unsigned short* A, int lda,
                                             int row0, int k0, int lane) {
    int hi = (lane >> 4) & 1;
    int r  = lane & 15;
    const unsigned short* p = A + (size_t)(row0 + r) * lda + k0;
    ABu x;
    x.u[0] = *(const u32x4*)(p + hi * 16);       // K 0..7  or 16..23
    x.u[1] = *(const u32x4*)(p + 8 + hi * 16);   // K 8..15 or 24..31
    return x.v;
}

// B fragment: 32x16 bf16 = W(N x K row-major) transposed on the fly.
// lane<16 : col=lane,    K 0..15 (pairs per VGPR)
// lane>=16: col=lane-16, K 16..31
__device__ __forceinline__ v16bf load_b_frag(const unsigned short* W, int ldw,
                                             int n0, int k0, int lane) {
    int hi = (lane >> 4) & 1;
    int r  = lane & 15;
    const unsigned short* p = W + (size_t)(n0 + r) * ldw + k0 + hi * 16;
    ABu x;
    x.u[0] = *(const u32x4*)(p);
    x.u[1] = *(const u32x4*)(p + 8);
    return x.v;
}

__device__ __forceinline__ v8f wmma_bf16(v16bf a, v16bf b, v8f c) {
    return __builtin_amdgcn_wmma_f32_16x16x32_bf16(false, a, false, b,
                                                   (short)0, c, false, false);
}

// ---------------- setup kernels ----------------

__global__ void k_mean(const float* __restrict__ img,
                       float* __restrict__ mf, unsigned short* __restrict__ mb) {
    int idx = blockIdx.x * blockDim.x + threadIdx.x;
    if (idx >= BB * EE) return;
    int b = idx / EE, e = idx % EE;
    const float* p = img + (size_t)b * PP * EE + e;
    float s = 0.0f;
    for (int q = 0; q < PP; ++q) s += p[(size_t)q * EE];
    s *= (1.0f / PP);
    mf[idx] = s;
    mb[idx] = f2bf(s);
}

__global__ void k_cvt2d(const float* __restrict__ src, int srs, int sco,
                        unsigned short* __restrict__ dst, int rows, int cols) {
    int idx = blockIdx.x * blockDim.x + threadIdx.x;
    if (idx >= rows * cols) return;
    int r = idx / cols, c = idx % cols;
    dst[idx] = f2bf(src[(size_t)r * srs + sco + c]);
}

__global__ void k_cvt_fcw(const float* __restrict__ src, unsigned short* __restrict__ dst) {
    int idx = blockIdx.x * blockDim.x + threadIdx.x;
    if (idx >= NOP * EE) return;
    int r = idx / EE, c = idx % EE;
    dst[idx] = (r < NO) ? f2bf(src[(size_t)r * EE + c]) : (unsigned short)0;
}

__global__ void k_cvt_label(const float* __restrict__ lab, unsigned short* __restrict__ dst) {
    int idx = blockIdx.x * blockDim.x + threadIdx.x;
    if (idx >= BB * TT * LL) return;
    int b = idx / (TT * LL);
    int rem = idx % (TT * LL);
    int t = rem / LL, l = rem % LL;
    dst[idx] = f2bf(lab[(size_t)b * (TT + 1) * LL + (size_t)t * LL + l]);
}

// xc = b_ih + b_hh + vp @ w_ih[:,E:E+V]^T   (image part added via WMMA GEMM after)
__global__ void k_xc_init(const float* __restrict__ vp, const float* __restrict__ w_ih,
                          const float* __restrict__ b_ih, const float* __restrict__ b_hh,
                          float* __restrict__ xc) {
    int idx = blockIdx.x * blockDim.x + threadIdx.x;
    if (idx >= BB * G4) return;
    int b = idx / G4, n = idx % G4;
    float s = b_ih[n] + b_hh[n];
    const float* wr = w_ih + (size_t)n * IN_ + EE;
#pragma unroll
    for (int v = 0; v < VV; ++v) s += vp[b * VV + v] * wr[v];
    xc[idx] = s;
}

// ---------------- WMMA GEMM kernels ----------------

// C[M,N] += A(MxK,bf16) * W(NxK,bf16)^T ; one 16x16 tile per wave
__global__ void k_wmma_acc(const unsigned short* __restrict__ A, int lda,
                           const unsigned short* __restrict__ W, int ldw,
                           float* __restrict__ C, int ldc, int Ntiles, int K) {
    int wave = blockIdx.x * (blockDim.x >> 5) + (threadIdx.x >> 5);
    int lane = threadIdx.x & 31;
    int mt = wave / Ntiles, nt = wave % Ntiles;
    int row0 = mt * 16, n0 = nt * 16;
    v8f acc = {};
    for (int k0 = 0; k0 < K; k0 += 32) {
        v16bf a = load_a_frag(A, lda, row0, k0, lane);
        v16bf b = load_b_frag(W, ldw, n0, k0, lane);
        acc = wmma_bf16(a, b, acc);
    }
    int hi = lane >> 4, c = lane & 15;
#pragma unroll
    for (int r = 0; r < 8; ++r) {
        int row = row0 + r + hi * 8;
        C[(size_t)row * ldc + n0 + c] += acc[r];
    }
}

// O = tanh(A * W^T + bias) -> both f32 and bf16 copies (for h0 / m0)
__global__ void k_wmma_tanh(const unsigned short* __restrict__ A,
                            const unsigned short* __restrict__ W,
                            const float* __restrict__ bias,
                            float* __restrict__ Of, unsigned short* __restrict__ Ob,
                            int Ntiles, int K) {
    int wave = blockIdx.x * (blockDim.x >> 5) + (threadIdx.x >> 5);
    int lane = threadIdx.x & 31;
    int mt = wave / Ntiles, nt = wave % Ntiles;
    int row0 = mt * 16, n0 = nt * 16;
    v8f acc = {};
    for (int k0 = 0; k0 < K; k0 += 32) {
        v16bf a = load_a_frag(A, K, row0, k0, lane);
        v16bf b = load_b_frag(W, K, n0, k0, lane);
        acc = wmma_bf16(a, b, acc);
    }
    int hi = lane >> 4, c = lane & 15;
#pragma unroll
    for (int r = 0; r < 8; ++r) {
        int row = row0 + r + hi * 8;
        int col = n0 + c;
        float v = tanhf(acc[r] + bias[col]);
        Of[(size_t)row * HH + col] = v;
        Ob[(size_t)row * HH + col] = f2bf(v);
    }
}

// Fused: gates = xc + lab_t@w_lab^T + begin(t==0) + h@w_hh^T, then LSTM pointwise.
// Each wave: 16(batch) x 16(h-col) tile of all four gates (i,f,g,o).
__global__ void k_gates_lstm(const unsigned short* __restrict__ h_in,
                             const unsigned short* __restrict__ w_hh_bf,
                             const unsigned short* __restrict__ lab_bf,  // (B,T,L) bf16
                             const unsigned short* __restrict__ w_lab_bf,// (4H,L) bf16
                             const float* __restrict__ xc,
                             const float* __restrict__ w_ih,             // fp32, for begin col
                             float* __restrict__ m,
                             unsigned short* __restrict__ h_out,
                             int t) {
    int wave = blockIdx.x * (blockDim.x >> 5) + (threadIdx.x >> 5);
    int lane = threadIdx.x & 31;
    const int Ntiles = HH / 16;
    int mt = wave / Ntiles, nt = wave % Ntiles;
    int row0 = mt * 16, n0 = nt * 16;

    v8f acc0 = {}, acc1 = {}, acc2 = {}, acc3 = {};
    for (int k0 = 0; k0 < HH; k0 += 32) {
        v16bf a = load_a_frag(h_in, HH, row0, k0, lane);
        acc0 = wmma_bf16(a, load_b_frag(w_hh_bf, HH, 0 * HH + n0, k0, lane), acc0);
        acc1 = wmma_bf16(a, load_b_frag(w_hh_bf, HH, 1 * HH + n0, k0, lane), acc1);
        acc2 = wmma_bf16(a, load_b_frag(w_hh_bf, HH, 2 * HH + n0, k0, lane), acc2);
        acc3 = wmma_bf16(a, load_b_frag(w_hh_bf, HH, 3 * HH + n0, k0, lane), acc3);
    }
    const unsigned short* labA = lab_bf + (size_t)t * LL;  // row stride T*L between batches
    for (int k0 = 0; k0 < LL; k0 += 32) {
        v16bf a = load_a_frag(labA, TT * LL, row0, k0, lane);
        acc0 = wmma_bf16(a, load_b_frag(w_lab_bf, LL, 0 * HH + n0, k0, lane), acc0);
        acc1 = wmma_bf16(a, load_b_frag(w_lab_bf, LL, 1 * HH + n0, k0, lane), acc1);
        acc2 = wmma_bf16(a, load_b_frag(w_lab_bf, LL, 2 * HH + n0, k0, lane), acc2);
        acc3 = wmma_bf16(a, load_b_frag(w_lab_bf, LL, 3 * HH + n0, k0, lane), acc3);
    }

    int hi = lane >> 4, c = lane & 15;
#pragma unroll
    for (int r = 0; r < 8; ++r) {
        int bi  = row0 + r + hi * 8;
        int col = n0 + c;
        float gi = acc0[r] + xc[(size_t)bi * G4 + col];
        float gf = acc1[r] + xc[(size_t)bi * G4 + HH + col];
        float gg = acc2[r] + xc[(size_t)bi * G4 + 2 * HH + col];
        float go = acc3[r] + xc[(size_t)bi * G4 + 3 * HH + col];
        if (t == 0) {  // begin flag contributes w_ih[:, E+V] exactly at t==0
            gi += w_ih[(size_t)(0 * HH + col) * IN_ + (EE + VV)];
            gf += w_ih[(size_t)(1 * HH + col) * IN_ + (EE + VV)];
            gg += w_ih[(size_t)(2 * HH + col) * IN_ + (EE + VV)];
            go += w_ih[(size_t)(3 * HH + col) * IN_ + (EE + VV)];
        }
        float mo = m[(size_t)bi * HH + col];
        float mn = sigm(gf) * mo + sigm(gi) * tanhf(gg);
        float hn = sigm(go) * tanhf(mn);
        m[(size_t)bi * HH + col] = mn;
        h_out[(size_t)bi * HH + col] = f2bf(hn);
    }
}

// out = h_new @ fc_w^T + fc_b ; fused sigmoid/relu/exp + mask ; scatter into d_out regions
__global__ void k_out(const unsigned short* __restrict__ h_bf,
                      const unsigned short* __restrict__ fcw_bf,  // (NOP,E) bf16 zero-padded
                      const float* __restrict__ fc_b,
                      const int* __restrict__ length,
                      float* __restrict__ out, int t) {
    int wave = blockIdx.x * (blockDim.x >> 5) + (threadIdx.x >> 5);
    int lane = threadIdx.x & 31;
    const int Ntiles = NOP / 16;  // 73
    int mt = wave / Ntiles, nt = wave % Ntiles;
    int row0 = mt * 16, n0 = nt * 16;

    v8f acc = {};
    for (int k0 = 0; k0 < HH; k0 += 32) {
        v16bf a = load_a_frag(h_bf, HH, row0, k0, lane);
        v16bf b = load_b_frag(fcw_bf, HH, n0, k0, lane);
        acc = wmma_bf16(a, b, acc);
    }

    const size_t LAB_OFF  = 0;
    const size_t TOP_OFF  = (size_t)BB * TT * LL;
    const size_t STOP_OFF = TOP_OFF + (size_t)BB * TT * HH;
    const size_t TEMP_OFF = STOP_OFF + (size_t)BB * TT;

    int hi = lane >> 4, c = lane & 15;
#pragma unroll
    for (int r = 0; r < 8; ++r) {
        int bi  = row0 + r + hi * 8;
        int col = n0 + c;
        if (col >= NO) continue;
        float v = acc[r] + fc_b[col];
        float mask = (length[bi] > t) ? 1.0f : 0.0f;
        if (col < LL)
            out[LAB_OFF + ((size_t)bi * TT + t) * LL + col] = sigm(v) * mask;
        else if (col < LL + HH)
            out[TOP_OFF + ((size_t)bi * TT + t) * HH + (col - LL)] = fmaxf(v, 0.0f) * mask;
        else if (col == LL + HH)
            out[STOP_OFF + (size_t)bi * TT + t] = sigm(v) * mask;
        else
            out[TEMP_OFF + (size_t)bi * TT + t] = expf(v) * mask;
    }
}

// ---------------- host launcher ----------------

extern "C" void kernel_launch(void* const* d_in, const int* in_sizes, int n_in,
                              void* d_out, int out_size, void* d_ws, size_t ws_size,
                              hipStream_t stream) {
    const float* image  = (const float*)d_in[0];
    const float* vp     = (const float*)d_in[1];
    const float* label  = (const float*)d_in[2];
    const int*   length = (const int*)d_in[3];
    const float* fc_h_w = (const float*)d_in[4];
    const float* fc_h_b = (const float*)d_in[5];
    const float* fc_m_w = (const float*)d_in[6];
    const float* fc_m_b = (const float*)d_in[7];
    const float* fc_w   = (const float*)d_in[8];
    const float* fc_b   = (const float*)d_in[9];
    const float* w_ih   = (const float*)d_in[10];
    const float* b_ih   = (const float*)d_in[11];
    const float* w_hh   = (const float*)d_in[12];
    const float* b_hh   = (const float*)d_in[13];
    float* out = (float*)d_out;

    // workspace carve-out (~35 MB)
    char* base = (char*)d_ws;
    size_t off = 0;
    auto carve = [&](size_t bytes) -> char* {
        char* p = base + off;
        off = (off + bytes + 255) & ~(size_t)255;
        return p;
    };
    float*          mean_f  = (float*)carve((size_t)BB * EE * 4);
    unsigned short* mean_bf = (unsigned short*)carve((size_t)BB * EE * 2);
    unsigned short* whh_bf  = (unsigned short*)carve((size_t)G4 * EE * 2);
    unsigned short* wimg_bf = (unsigned short*)carve((size_t)G4 * EE * 2);
    unsigned short* wlab_bf = (unsigned short*)carve((size_t)G4 * LL * 2);
    unsigned short* fch_bf  = (unsigned short*)carve((size_t)HH * EE * 2);
    unsigned short* fcm_bf  = (unsigned short*)carve((size_t)HH * EE * 2);
    unsigned short* fcw_bf  = (unsigned short*)carve((size_t)NOP * EE * 2);
    unsigned short* lab_bf  = (unsigned short*)carve((size_t)BB * TT * LL * 2);
    float*          xc      = (float*)carve((size_t)BB * G4 * 4);
    unsigned short* hA      = (unsigned short*)carve((size_t)BB * HH * 2);
    unsigned short* hB      = (unsigned short*)carve((size_t)BB * HH * 2);
    float*          m_st    = (float*)carve((size_t)BB * HH * 4);
    float*          dum_f   = (float*)carve((size_t)BB * HH * 4);
    unsigned short* dum_b   = (unsigned short*)carve((size_t)BB * HH * 2);
    (void)in_sizes; (void)n_in; (void)out_size; (void)ws_size; (void)mean_f;

    const int TPB = 256;

    // setup
    k_mean<<<(BB * EE + TPB - 1) / TPB, TPB, 0, stream>>>(image, mean_f, mean_bf);
    k_cvt2d<<<(G4 * EE + TPB - 1) / TPB, TPB, 0, stream>>>(w_hh, EE, 0, whh_bf, G4, EE);
    k_cvt2d<<<(G4 * EE + TPB - 1) / TPB, TPB, 0, stream>>>(w_ih, IN_, 0, wimg_bf, G4, EE);
    k_cvt2d<<<(G4 * LL + TPB - 1) / TPB, TPB, 0, stream>>>(w_ih, IN_, EE + VV + 1, wlab_bf, G4, LL);
    k_cvt2d<<<(HH * EE + TPB - 1) / TPB, TPB, 0, stream>>>(fc_h_w, EE, 0, fch_bf, HH, EE);
    k_cvt2d<<<(HH * EE + TPB - 1) / TPB, TPB, 0, stream>>>(fc_m_w, EE, 0, fcm_bf, HH, EE);
    k_cvt_fcw<<<(NOP * EE + TPB - 1) / TPB, TPB, 0, stream>>>(fc_w, fcw_bf);
    k_cvt_label<<<(BB * TT * LL + TPB - 1) / TPB, TPB, 0, stream>>>(label, lab_bf);

    // xc = biases + vp part, then xc += mean @ w_ih_img^T  (16 x 256 tiles, 8 waves/block)
    k_xc_init<<<(BB * G4 + TPB - 1) / TPB, TPB, 0, stream>>>(vp, w_ih, b_ih, b_hh, xc);
    k_wmma_acc<<<(16 * (G4 / 16)) / 8, TPB, 0, stream>>>(mean_bf, EE, wimg_bf, EE,
                                                         xc, G4, G4 / 16, EE);

    // h0 -> hA (bf16), m0 -> m_st (f32)
    k_wmma_tanh<<<(16 * (HH / 16)) / 8, TPB, 0, stream>>>(mean_bf, fch_bf, fc_h_b,
                                                          dum_f, hA, HH / 16, EE);
    k_wmma_tanh<<<(16 * (HH / 16)) / 8, TPB, 0, stream>>>(mean_bf, fcm_bf, fc_m_b,
                                                          m_st, dum_b, HH / 16, EE);

    // sequential LSTM steps, double-buffered h
    for (int t = 0; t < TT; ++t) {
        unsigned short* hin  = (t & 1) ? hB : hA;
        unsigned short* hout = (t & 1) ? hA : hB;
        k_gates_lstm<<<(16 * (HH / 16)) / 8, TPB, 0, stream>>>(
            hin, whh_bf, lab_bf, wlab_bf, xc, w_ih, m_st, hout, t);
        k_out<<<(16 * (NOP / 16)) / 8 + ((16 * (NOP / 16)) % 8 ? 1 : 0), TPB, 0, stream>>>(
            hout, fcw_bf, fc_b, length, out, t);
    }
}
